// WeightEnforcer_57595511439510
// MI455X (gfx1250) — matrix operand order repaired and checked by reading, not compiled, not run on hardware
//
#include <hip/hip_runtime.h>
#include <hip/hip_bf16.h>

typedef __attribute__((ext_vector_type(16))) _Float16 v16h;
typedef __attribute__((ext_vector_type(8)))  float    v8f;

namespace {
constexpr int kH = 256, kW = 256;
constexpr int kBC = 32;          // b*c
constexpr int kN  = 32;          // n_psfs
constexpr int kD  = 64;          // low-res weight grid (h/4)
constexpr int kPix  = 8;         // pixels per block (one per wave)
constexpr int kSlab = 40;        // padded x slab width (39 valid: w0-15 .. w0+23)
constexpr int kWyPad = 36;       // padded k-extent of y-lerped weights (16B-aligned rows)
}

union Frag16 { v16h v; _Float16 h[16]; };

__device__ __forceinline__ int clampi(int v, int lo, int hi) {
  return v < lo ? lo : (v > hi ? hi : v);
}

__global__ __launch_bounds__(256)
void we_banded_wmma(const float* __restrict__ pw,   // [32, 64*64, 31]
                    const float* __restrict__ x,    // [32, 256, 256]
                    float* __restrict__ out)        // [32, 32, 65536]
{
  __shared__ _Float16 Xs[kBC][kSlab];        // x row slab, f16
  __shared__ float    Wy[4][kN][kWyPad];     // y-lerped weights for 4 columns
  __shared__ float    Ds[kBC][kN][kPix];     // D staging for coalesced stores

  const int t    = threadIdx.x;
  const int lane = t & 31;
  const int wv   = t >> 5;

  const int blk = blockIdx.x;                // 8192 blocks
  const int row = blk >> 5;                  // 32 blocks per image row
  const int w0  = (blk & 31) * kPix;

  // ---- stage x[bc, row, w0-15 .. w0+23] -> Xs (zero pad outside width) ----
  for (int i = t; i < kBC * kSlab; i += 256) {
    const int m  = i / kSlab;
    const int c  = i % kSlab;
    const int gw = w0 - 15 + c;
    float v = 0.0f;
    if (c < 39 && gw >= 0 && gw < kW)
      v = x[(size_t)m * (kH * kW) + (size_t)row * kW + gw];
    Xs[m][c] = (_Float16)v;
  }

  // ---- bilinear source coords (align_corners=False, scale 4) ----
  const float sy  = row * 0.25f - 0.375f;
  const float y0f = floorf(sy);
  const float ty  = sy - y0f;
  const int y0 = clampi((int)y0f,     0, kD - 1);
  const int y1 = clampi((int)y0f + 1, 0, kD - 1);

  // ---- cooperatively stage y-lerped weights for the 4 columns this block
  //      touches: Wy[c][n][k] = (1-ty)*pw[n,y0,col,k] + ty*pw[n,y1,col,k]
  //      (k >= 31 stored as 0 -> zero-pads the WMMA K=31 column) ----
  const int xb0 = (w0 >> 2) - 1;
  for (int i = t; i < 4 * kN * 32; i += 256) {
    const int c = i >> 10;                   // 0..3
    const int n = (i >> 5) & 31;
    const int k = i & 31;
    const int col = clampi(xb0 + c, 0, kD - 1);
    float v = 0.0f;
    if (k < 31) {
      const float* base = pw + (size_t)n * (kD * kD * 31) + (size_t)col * 31 + k;
      v = (1.0f - ty) * base[(size_t)y0 * (kD * 31)] +
                   ty * base[(size_t)y1 * (kD * 31)];
    }
    Wy[c][n][k] = v;
    if (k >= 27) Wy[c][n][k + 4] = 0.0f;     // fill pad tail 32..35
  }
  __syncthreads();

  // ---- this wave's pixel ----
  const int pl = wv;                         // wx - w0
  const int wx = w0 + pl;
  const float sx  = wx * 0.25f - 0.375f;
  const float x0f = floorf(sx);
  const float tx  = sx - x0f;
  const int c0 = (int)x0f - xb0;             // 0..2 -> Wy column pair (c0, c0+1)
  const float f1 = tx, f0 = 1.0f - tx;

  // ---- A fragments (contiguous k = 16h + s, same order as B) ----
  // lanes 0-15: row m=lane,    K = 0..15
  // lanes16-31: row m=lane-16, K = 16..31
  const int am   = lane & 15;
  const int kofs = (lane < 16) ? 0 : 16;
  Frag16 A0, A1;
#pragma unroll
  for (int j = 0; j < 8; ++j) {
    A0.h[j]     = Xs[am][pl + kofs + j];
    A0.h[j + 8] = Xs[am][pl + kofs + 8 + j];
    A1.h[j]     = Xs[am + 16][pl + kofs + j];
    A1.h[j + 8] = Xs[am + 16][pl + kofs + 8 + j];
  }

  // ---- B fragments: one x-lerp from LDS (b128 reads) ----
  // lanes 0-15: col n=lane, K=0..15 ; lanes 16-31: col n=lane-16, K=16..31
  const int bn = lane & 15;
  const int q  = (lane < 16) ? 0 : 4;        // float4 chunk index of k0
  Frag16 B0, B1;
#pragma unroll
  for (int nt = 0; nt < 2; ++nt) {
    Frag16& B = nt ? B1 : B0;
    const int n = bn + nt * 16;
    const float4* wp0 = (const float4*)&Wy[c0][n][0];
    const float4* wp1 = (const float4*)&Wy[c0 + 1][n][0];
    float4 a4[4], b4[4];
#pragma unroll
    for (int u = 0; u < 4; ++u) { a4[u] = wp0[q + u]; b4[u] = wp1[q + u]; }
    const float* fa = (const float*)a4;
    const float* fb = (const float*)b4;
#pragma unroll
    for (int j = 0; j < 16; ++j)
      B.h[j] = (_Float16)(f0 * fa[j] + f1 * fb[j]);
  }

  // ---- 4 WMMAs: full 32x32x32 for this pixel ----
  v8f a00 = {}, a01 = {}, a10 = {}, a11 = {};
  a00 = __builtin_amdgcn_wmma_f32_16x16x32_f16(false, A0.v, false, B0.v,
                                               (short)0, a00, false, false);
  a01 = __builtin_amdgcn_wmma_f32_16x16x32_f16(false, A0.v, false, B1.v,
                                               (short)0, a01, false, false);
  a10 = __builtin_amdgcn_wmma_f32_16x16x32_f16(false, A1.v, false, B0.v,
                                               (short)0, a10, false, false);
  a11 = __builtin_amdgcn_wmma_f32_16x16x32_f16(false, A1.v, false, B1.v,
                                               (short)0, a11, false, false);

  // ---- scatter D fragments to LDS (C/D f32 16x16 lane layout) ----
  // VGPR r: lanes 0-15 -> M=r, N=lane ; lanes 16-31 -> M=r+8, N=lane-16
  const int dn  = lane & 15;
  const int mhi = (lane < 16) ? 0 : 8;
#pragma unroll
  for (int r = 0; r < 8; ++r) {
    Ds[r + mhi][dn][pl]           = a00[r];
    Ds[r + mhi][16 + dn][pl]      = a01[r];
    Ds[16 + r + mhi][dn][pl]      = a10[r];
    Ds[16 + r + mhi][16 + dn][pl] = a11[r];
  }
  __syncthreads();

  // ---- coalesced output: out[m][n][row*256 + w0 .. +7], 16B-aligned b128 ----
  const size_t pbase = (size_t)row * kW + w0;
#pragma unroll
  for (int i = 0; i < 8; ++i) {
    const int id = t + i * 256;              // 0..2047
    const int mn = id >> 1;                  // (m*32+n), 0..1023
    const int c4 = id & 1;                   // float4 chunk within 8 pixels
    const float4 val = *(const float4*)&Ds[mn >> 5][mn & 31][c4 * 4];
    *(float4*)(out + (size_t)mn * (kH * kW) + pbase + c4 * 4) = val;
  }
}

extern "C" void kernel_launch(void* const* d_in, const int* in_sizes, int n_in,
                              void* d_out, int out_size, void* d_ws, size_t ws_size,
                              hipStream_t stream) {
  (void)in_sizes; (void)n_in; (void)out_size; (void)d_ws; (void)ws_size;
  const float* pw = (const float*)d_in[0];   // [32, 4096, 31]
  const float* x  = (const float*)d_in[1];   // [4, 8, 256, 256]
  float* out = (float*)d_out;                // [32, 32, 256, 256]
  dim3 grid((kH * kW) / kPix);               // 8192 blocks
  we_banded_wmma<<<grid, 256, 0, stream>>>(pw, x, out);
}
